// CapsAlexNet_FLOWER102_87488483820218
// MI455X (gfx1250) — matrix operand order repaired and checked by reading, not compiled, not run on hardware
//
#include <hip/hip_runtime.h>
#include <math.h>

typedef __attribute__((ext_vector_type(16))) _Float16 v16h;
typedef __attribute__((ext_vector_type(8)))  float    v8f;

// ---------------------------------------------------------------------------
// elementwise helpers
// ---------------------------------------------------------------------------
__global__ void k_zero_f32(float* p, int n) {
  int i = blockIdx.x * blockDim.x + threadIdx.x;
  if (i < n) p[i] = 0.f;
}

__global__ void k_f32_to_f16(const float* __restrict__ in, _Float16* __restrict__ out, int n) {
  int i = blockIdx.x * blockDim.x + threadIdx.x;
  if (i < n) out[i] = (_Float16)in[i];
}

// ---------------------------------------------------------------------------
// direct NCHW conv (+bias, optional relu) — used for conv1 and conv3 (small)
// ---------------------------------------------------------------------------
__global__ void k_conv_direct(const float* __restrict__ in, const float* __restrict__ w,
                              const float* __restrict__ bias, float* __restrict__ out,
                              int Bn, int C, int H, int W, int OC, int KH, int KW,
                              int stride, int pad, int OH, int OW, int relu)
{
  int idx = blockIdx.x * blockDim.x + threadIdx.x;
  int total = Bn * OC * OH * OW;
  if (idx >= total) return;
  int ox = idx % OW; int t = idx / OW;
  int oy = t % OH;   t /= OH;
  int oc = t % OC;   int b = t / OC;
  float acc = bias[oc];
  for (int c = 0; c < C; ++c)
    for (int ky = 0; ky < KH; ++ky) {
      int iy = oy * stride - pad + ky;
      if (iy < 0 || iy >= H) continue;
      for (int kx = 0; kx < KW; ++kx) {
        int ix = ox * stride - pad + kx;
        if (ix < 0 || ix >= W) continue;
        acc += in[((size_t)(b * C + c) * H + iy) * W + ix] *
               w[((size_t)(oc * C + c) * KH + ky) * KW + kx];
      }
    }
  if (relu) acc = fmaxf(acc, 0.f);
  out[((size_t)(b * OC + oc) * OH + oy) * OW + ox] = acc;
}

// ---------------------------------------------------------------------------
// VALID max pool (NCHW)
// ---------------------------------------------------------------------------
__global__ void k_maxpool(const float* __restrict__ in, float* __restrict__ out,
                          int Bn, int C, int H, int W, int k, int s, int OH, int OW)
{
  int idx = blockIdx.x * blockDim.x + threadIdx.x;
  int total = Bn * C * OH * OW;
  if (idx >= total) return;
  int ox = idx % OW; int t = idx / OW;
  int oy = t % OH;   t /= OH;
  int c = t % C;     int b = t / C;
  float m = -INFINITY;
  for (int ky = 0; ky < k; ++ky) {
    int iy = oy * s + ky; if (iy >= H) break;
    for (int kx = 0; kx < k; ++kx) {
      int ix = ox * s + kx; if (ix >= W) break;
      m = fmaxf(m, in[((size_t)(b * C + c) * H + iy) * W + ix]);
    }
  }
  out[idx] = m;
}

// ---------------------------------------------------------------------------
// im2col into f16 patch matrix  [M = B*OH*OW,  K = C*KH*KW]
// ---------------------------------------------------------------------------
__global__ void k_im2col_f16(const float* __restrict__ in, _Float16* __restrict__ out,
                             int Bn, int C, int H, int W, int KH, int KW,
                             int stride, int pad, int OH, int OW)
{
  int K = C * KH * KW;
  int idx = blockIdx.x * blockDim.x + threadIdx.x;
  int total = Bn * OH * OW * K;
  if (idx >= total) return;
  int k = idx % K; int m = idx / K;
  int ox = m % OW; int t = m / OW;
  int oy = t % OH; int b = t / OH;
  int c = k / (KH * KW); int r = k % (KH * KW);
  int ky = r / KW; int kx = r % KW;
  int iy = oy * stride - pad + ky;
  int ix = ox * stride - pad + kx;
  float v = 0.f;
  if (iy >= 0 && iy < H && ix >= 0 && ix < W)
    v = in[((size_t)(b * C + c) * H + iy) * W + ix];
  out[idx] = (_Float16)v;
}

// ---------------------------------------------------------------------------
// WMMA GEMM:  D[M,N] = act(A[M,K](f16) @ B[N,K]^T(f32, converted inline) + bias)
//   - one 16x16 output tile per wave, 4 waves/block span 64 columns
//   - ohw>0: scatter store to NCHW:  m=(b,pix) -> out[(b*cout + n)*ohw + pix]
//   - Dh: optional fused f16 copy of the output (feeds next GEMM's A)
// A-operand layout (16-bit 16x32): lane holds row (lane&15); halves 0..7 are
// K = 8*(lane>>4)+j, halves 8..15 are K+16.  B-operand: lane holds column
// (lane&15); halves use the same K order as A.  D: N=lane&15, M=r+8*(lane>>4).
// ---------------------------------------------------------------------------
__global__ void k_gemm_wmma(const _Float16* __restrict__ A, const float* __restrict__ Bw,
                            const float* __restrict__ bias, float* __restrict__ D,
                            _Float16* __restrict__ Dh,
                            int M, int N, int K, int relu, int ohw, int cout)
{
  int lane = threadIdx.x & 31;
  int wave = threadIdx.x >> 5;
  int tileM = blockIdx.x << 4;
  int tileN = (blockIdx.y << 6) + (wave << 4);
  if (tileN >= N) return;                 // wave-uniform: EXEC stays all-ones
  int l = lane & 15;
  int h = lane >> 4;
  int row = tileM + l;
  int col = tileN + l;
  bool rok = row < M;
  bool cok = col < N;
  v8f acc = {};
  for (int kk = 0; kk < K; kk += 32) {
    v16h a = {};
    v16h bmat = {};
    if (rok) {
      const _Float16* ap = A + (size_t)row * K + kk + h * 8;
      #pragma unroll
      for (int j = 0; j < 8; ++j) { a[j] = ap[j]; a[8 + j] = ap[16 + j]; }
    }
    if (cok) {
      const float* bp = Bw + (size_t)col * K + kk + h * 8;
      #pragma unroll
      for (int t = 0; t < 8; ++t) { bmat[t] = (_Float16)bp[t]; bmat[8 + t] = (_Float16)bp[16 + t]; }
    }
    acc = __builtin_amdgcn_wmma_f32_16x16x32_f16(false, a, false, bmat,
                                                 (short)0, acc, false, false);
  }
  int n = tileN + l;
  if (n >= N) return;
  float bv = bias ? bias[n] : 0.f;
  #pragma unroll
  for (int r = 0; r < 8; ++r) {
    int m = tileM + r + h * 8;
    if (m >= M) continue;
    float val = acc[r] + bv;
    if (relu) val = fmaxf(val, 0.f);
    size_t oidx = (ohw > 0) ? ((size_t)(m / ohw) * cout + n) * ohw + (m % ohw)
                            : (size_t)m * N + n;
    D[oidx] = val;
    if (Dh) Dh[oidx] = (_Float16)val;
  }
}

// ---------------------------------------------------------------------------
// primary-caps squash: p [B, 2592, 8] (flat NCHW) -> u f16 [B,2592,8]
// ---------------------------------------------------------------------------
__global__ void k_squash_primary(const float* __restrict__ p, _Float16* __restrict__ u,
                                 int Bn, int ncaps)
{
  int idx = blockIdx.x * blockDim.x + threadIdx.x;
  if (idx >= Bn * ncaps) return;
  const float* s = p + (size_t)idx * 8;
  float n2 = 0.f;
  #pragma unroll
  for (int k = 0; k < 8; ++k) n2 += s[k] * s[k];
  float sc = (n2 / (1.f + n2)) * rsqrtf(n2 + 1e-8f);
  _Float16* o = u + (size_t)idx * 8;
  #pragma unroll
  for (int k = 0; k < 8; ++k) o[k] = (_Float16)(s[k] * sc);
}

// ---------------------------------------------------------------------------
// x_hat[b,o,i,d] = sum_k Wcap[o,i,d,k] * u[b,i,k]
// one 16(B) x 16(Dout) x 8(K, zero-padded to 32) WMMA per (o,i); one wave/block
// ---------------------------------------------------------------------------
__global__ void k_xhat_wmma(const _Float16* __restrict__ u, const _Float16* __restrict__ Wh,
                            _Float16* __restrict__ xhat, int I, int O)
{
  int i = blockIdx.x;
  int o = blockIdx.y;
  int lane = threadIdx.x & 31;
  int l = lane & 15;
  int h = lane >> 4;
  v16h a = {};
  v16h bmat = {};
  if (h == 0) {                       // only K=0..7 are non-zero
    const _Float16* up = u + ((size_t)l * I + i) * 8;       // row = batch l
    #pragma unroll
    for (int j = 0; j < 8; ++j) a[j] = up[j];
    const _Float16* wp = Wh + (((size_t)o * I + i) * 16 + l) * 8;  // col = dout l
    #pragma unroll
    for (int t = 0; t < 8; ++t) bmat[t] = wp[t];
  }
  v8f acc = {};
  acc = __builtin_amdgcn_wmma_f32_16x16x32_f16(false, a, false, bmat,
                                               (short)0, acc, false, false);
  #pragma unroll
  for (int r = 0; r < 8; ++r) {
    int b = r + h * 8;
    xhat[(((size_t)b * O + o) * I + i) * 16 + l] = (_Float16)acc[r];
  }
}

// ---------------------------------------------------------------------------
// routing: softmax of b_logits over o (per b,i)
// ---------------------------------------------------------------------------
__global__ void k_route_softmax(const float* __restrict__ blog, float* __restrict__ c,
                                int Bn, int O, int I)
{
  int idx = blockIdx.x * blockDim.x + threadIdx.x;
  if (idx >= Bn * I) return;
  int b = idx / I; int i = idx % I;
  size_t base = (size_t)b * O * I + i;
  float mx = -INFINITY;
  for (int o = 0; o < O; ++o) mx = fmaxf(mx, blog[base + (size_t)o * I]);
  float se = 0.f;
  for (int o = 0; o < O; ++o) {
    float e = __expf(blog[base + (size_t)o * I] - mx);
    c[base + (size_t)o * I] = e;
    se += e;
  }
  float inv = 1.f / se;
  for (int o = 0; o < O; ++o) c[base + (size_t)o * I] *= inv;
}

// ---------------------------------------------------------------------------
// routing: v[b,o,:] = squash( sum_i c[b,o,i] * xhat[b,o,i,:] );  one block per (b,o)
// ---------------------------------------------------------------------------
__global__ void k_route_reduce_squash(const float* __restrict__ c,
                                      const _Float16* __restrict__ xhat,
                                      float* __restrict__ v, int I)
{
  int bo = blockIdx.x;
  int d = threadIdx.x;   // 0..15 (capsule dim)
  int t = threadIdx.y;   // 0..15 (i-strided partial)
  size_t base = (size_t)bo * I;
  float p = 0.f;
  for (int i = t; i < I; i += 16)
    p += c[base + i] * (float)xhat[(base + i) * 16 + d];
  __shared__ float red[16][17];
  red[t][d] = p;
  __syncthreads();
  if (t == 0) {
    float s = 0.f;
    for (int j = 0; j < 16; ++j) s += red[j][d];
    red[0][d] = s;
  }
  __syncthreads();
  if (t == 0) {
    float n2 = 0.f;
    for (int j = 0; j < 16; ++j) n2 += red[0][j] * red[0][j];
    float sc = (n2 / (1.f + n2)) * rsqrtf(n2 + 1e-8f);
    v[bo * 16 + d] = red[0][d] * sc;
  }
}

// ---------------------------------------------------------------------------
// routing: b_logits[b,o,i] += <v[b,o,:], xhat[b,o,i,:]>
// ---------------------------------------------------------------------------
__global__ void k_route_update(float* __restrict__ blog, const float* __restrict__ v,
                               const _Float16* __restrict__ xhat, int I, int total)
{
  int idx = blockIdx.x * blockDim.x + threadIdx.x;
  if (idx >= total) return;
  int bo = idx / I;
  const float* vv = v + (size_t)bo * 16;
  const _Float16* xp = xhat + (size_t)idx * 16;
  float dot = 0.f;
  #pragma unroll
  for (int d = 0; d < 16; ++d) dot += vv[d] * (float)xp[d];
  blog[idx] += dot;
}

// ---------------------------------------------------------------------------
// host-side launch
// ---------------------------------------------------------------------------
extern "C" void kernel_launch(void* const* d_in, const int* in_sizes, int n_in,
                              void* d_out, int out_size, void* d_ws, size_t ws_size,
                              hipStream_t stream)
{
  (void)in_sizes; (void)n_in; (void)out_size; (void)ws_size;

  const float* x    = (const float*)d_in[0];
  const float* w1   = (const float*)d_in[1];
  const float* b1   = (const float*)d_in[2];
  const float* w2   = (const float*)d_in[3];
  const float* b2   = (const float*)d_in[4];
  const float* wp   = (const float*)d_in[5];
  const float* bp   = (const float*)d_in[6];
  const float* Wcap = (const float*)d_in[7];
  const float* w3   = (const float*)d_in[8];
  const float* b3   = (const float*)d_in[9];
  const float* fw1  = (const float*)d_in[10];
  const float* fb1  = (const float*)d_in[11];
  const float* fw2  = (const float*)d_in[12];
  const float* fb2  = (const float*)d_in[13];
  const float* fw3  = (const float*)d_in[14];
  const float* fb3  = (const float*)d_in[15];

  const int B = 16, O = 102, I = 2592;

  char* base = (char*)d_ws;
  size_t off = 0;
  auto carve = [&](size_t bytes) -> void* {
    void* p = base + off;
    off = (off + bytes + 255) & ~(size_t)255;
    return p;
  };

  float*    a1    = (float*)   carve(sizeof(float)    * (size_t)B * 96 * 51 * 51);
  float*    p1    = (float*)   carve(sizeof(float)    * (size_t)B * 96 * 25 * 25);
  _Float16* A2    = (_Float16*)carve(sizeof(_Float16) * (size_t)10000 * 2400);
  float*    h2c   = (float*)   carve(sizeof(float)    * (size_t)B * 256 * 25 * 25);
  float*    p2    = (float*)   carve(sizeof(float)    * (size_t)B * 256 * 12 * 12);
  _Float16* Ap    = (_Float16*)carve(sizeof(_Float16) * (size_t)1296 * 4096);
  float*    pcaps = (float*)   carve(sizeof(float)    * (size_t)B * 256 * 81);
  _Float16* u16   = (_Float16*)carve(sizeof(_Float16) * (size_t)B * I * 8);
  _Float16* Wh    = (_Float16*)carve(sizeof(_Float16) * (size_t)O * I * 16 * 8);
  _Float16* xhat  = (_Float16*)carve(sizeof(_Float16) * (size_t)B * O * I * 16);
  float*    blog  = (float*)   carve(sizeof(float)    * (size_t)B * O * I);
  float*    cbuf  = (float*)   carve(sizeof(float)    * (size_t)B * O * I);
  float*    vbuf  = (float*)   carve(sizeof(float)    * (size_t)B * O * 16);
  float*    h3    = (float*)   carve(sizeof(float)    * (size_t)B * 256 * 50 * 7);
  float*    fbuf  = (float*)   carve(sizeof(float)    * (size_t)B * 18432);
  _Float16* fA    = (_Float16*)carve(sizeof(_Float16) * (size_t)B * 18432);
  float*    fc1   = (float*)   carve(sizeof(float)    * (size_t)B * 4096);
  _Float16* fc1h  = (_Float16*)carve(sizeof(_Float16) * (size_t)B * 4096);
  float*    fc2   = (float*)   carve(sizeof(float)    * (size_t)B * 4096);
  _Float16* fc2h  = (_Float16*)carve(sizeof(_Float16) * (size_t)B * 4096);

  auto blks = [](long long n) { return dim3((unsigned)((n + 255) / 256)); };

  // ---- conv1 + relu: [16,3,211,211] -> [16,96,51,51]
  k_conv_direct<<<blks((long long)B * 96 * 51 * 51), 256, 0, stream>>>(
      x, w1, b1, a1, B, 3, 211, 211, 96, 11, 11, 4, 0, 51, 51, 1);
  // ---- pool1 -> [16,96,25,25]
  k_maxpool<<<blks((long long)B * 96 * 25 * 25), 256, 0, stream>>>(
      a1, p1, B, 96, 51, 51, 3, 2, 25, 25);

  // ---- conv2 via im2col + WMMA GEMM: M=10000, N=256, K=2400
  k_im2col_f16<<<blks((long long)10000 * 2400), 256, 0, stream>>>(
      p1, A2, B, 96, 25, 25, 5, 5, 1, 2, 25, 25);
  k_gemm_wmma<<<dim3(10000 / 16, 256 / 64), 128, 0, stream>>>(
      A2, w2, b2, h2c, nullptr, 10000, 256, 2400, 1, 625, 256);
  // ---- pool2 -> [16,256,12,12]
  k_maxpool<<<blks((long long)B * 256 * 12 * 12), 256, 0, stream>>>(
      h2c, p2, B, 256, 25, 25, 3, 2, 12, 12);

  // ---- primary caps conv via im2col + WMMA GEMM: M=1296, N=256, K=4096
  k_im2col_f16<<<blks((long long)1296 * 4096), 256, 0, stream>>>(
      p2, Ap, B, 256, 12, 12, 4, 4, 1, 0, 9, 9);
  k_gemm_wmma<<<dim3(1296 / 16, 256 / 64), 128, 0, stream>>>(
      Ap, wp, bp, pcaps, nullptr, 1296, 256, 4096, 0, 81, 256);

  // ---- squash -> u f16 [16,2592,8]
  k_squash_primary<<<blks((long long)B * I), 256, 0, stream>>>(pcaps, u16, B, I);

  // ---- Wcap -> f16, then x_hat via per-(o,i) WMMA
  k_f32_to_f16<<<blks((long long)O * I * 16 * 8), 256, 0, stream>>>(
      Wcap, Wh, O * I * 16 * 8);
  k_xhat_wmma<<<dim3(I, O), 32, 0, stream>>>(u16, Wh, xhat, I, O);

  // ---- dynamic routing (3 iterations)
  int blog_n = B * O * I;
  k_zero_f32<<<blks(blog_n), 256, 0, stream>>>(blog, blog_n);
  for (int it = 0; it < 3; ++it) {
    k_route_softmax<<<blks((long long)B * I), 256, 0, stream>>>(blog, cbuf, B, O, I);
    k_route_reduce_squash<<<B * O, dim3(16, 16), 0, stream>>>(cbuf, xhat, vbuf, I);
    if (it < 2)
      k_route_update<<<blks(blog_n), 256, 0, stream>>>(blog, vbuf, xhat, I, blog_n);
  }

  // ---- conv3 + relu on capsule map [16,1,102,16] -> [16,256,50,7]
  k_conv_direct<<<blks((long long)B * 256 * 50 * 7), 256, 0, stream>>>(
      vbuf, w3, b3, h3, B, 1, 102, 16, 256, 3, 3, 2, 0, 50, 7, 1);
  // ---- pool3 -> [16,256,24,3] == f [16,18432]
  k_maxpool<<<blks((long long)B * 256 * 24 * 3), 256, 0, stream>>>(
      h3, fbuf, B, 256, 50, 7, 3, 2, 24, 3);
  k_f32_to_f16<<<blks((long long)B * 18432), 256, 0, stream>>>(fbuf, fA, B * 18432);

  // ---- FC stack on WMMA (M=16, weight-streaming bound)
  k_gemm_wmma<<<dim3(1, 4096 / 64), 128, 0, stream>>>(
      fA, fw1, fb1, fc1, fc1h, 16, 4096, 18432, 1, 0, 0);
  k_gemm_wmma<<<dim3(1, 4096 / 64), 128, 0, stream>>>(
      fc1h, fw2, fb2, fc2, fc2h, 16, 4096, 4096, 1, 0, 0);
  k_gemm_wmma<<<dim3(1, 2), 128, 0, stream>>>(
      fc2h, fw3, fb3, (float*)d_out, nullptr, 16, 102, 4096, 0, 0, 0);
}
